// GraphAttentionLayerClass_79147657331216
// MI455X (gfx1250) — matrix operand order repaired and checked by reading, not compiled, not run on hardware
//
#include <hip/hip_runtime.h>
#include <hip/hip_bf16.h>

typedef __attribute__((ext_vector_type(16))) __bf16 v16bf;
typedef __attribute__((ext_vector_type(8)))  __bf16 v8bf;
typedef __attribute__((ext_vector_type(8)))  float  v8f;

#define N_NODES 8192
#define FDIM    512
#define ALPHA   0.2f
#define MASKV   -9000000000000000.0f

#define BM 128
#define BN 64
#define BK 32
#define LDS_STRIDE 40   // 32 + 8 pad (bf16 elems) to spread LDS banks

__device__ __forceinline__ float waveReduceSum(float v) {
  #pragma unroll
  for (int o = 16; o > 0; o >>= 1) v += __shfl_xor(v, o, 32);
  return v;
}
__device__ __forceinline__ float waveReduceMax(float v) {
  #pragma unroll
  for (int o = 16; o > 0; o >>= 1) v = fmaxf(v, __shfl_xor(v, o, 32));
  return v;
}

// ---------------------------------------------------------------------------
// K1: Wh = h (8192x512 f32) @ W (512x512 f32), bf16 WMMA, f32 accumulate.
// Also emits a bf16 copy of Wh for the second GEMM's B operand.
// ---------------------------------------------------------------------------
__global__ __launch_bounds__(256) void k_gemm_wh(const float* __restrict__ H,
                                                 const float* __restrict__ W,
                                                 float* __restrict__ Wh,
                                                 __bf16* __restrict__ Whb) {
  __shared__ __bf16 As[BM][LDS_STRIDE];
  __shared__ __bf16 Bt[BN][LDS_STRIDE];

  const int tid  = threadIdx.x;
  const int m0   = blockIdx.x * BM;
  const int n0   = blockIdx.y * BN;
  const int wv   = tid >> 5;
  const int lane = tid & 31;
  const int wm   = (wv & 3) * 32;   // 4 waves tile M
  const int wn   = (wv >> 2) * 32;  // 2 waves tile N
  const int l16    = lane & 15;
  const int kA     = (lane >> 4) * 8;   // A frag K base per half-wave
  const int kB     = (lane >> 4) * 8;   // B frag K base per half-wave

  // A staging: each thread loads 16 contiguous floats (half a 32-wide row)
  const int arow = tid >> 1;
  const int acol = (tid & 1) * 16;
  // B staging: each thread loads 8 contiguous floats of W
  const int brow = tid >> 3;
  const int bcol = (tid & 7) * 8;

  v8f acc[2][2] = {};

  for (int k0 = 0; k0 < FDIM; k0 += BK) {
    const float* ap = H + (size_t)(m0 + arow) * FDIM + (k0 + acol);
    #pragma unroll
    for (int t = 0; t < 4; ++t) {
      float4 f = ((const float4*)ap)[t];
      As[arow][acol + t * 4 + 0] = (__bf16)f.x;
      As[arow][acol + t * 4 + 1] = (__bf16)f.y;
      As[arow][acol + t * 4 + 2] = (__bf16)f.z;
      As[arow][acol + t * 4 + 3] = (__bf16)f.w;
    }
    const float* bp = W + (size_t)(k0 + brow) * FDIM + (n0 + bcol);
    #pragma unroll
    for (int t = 0; t < 2; ++t) {
      float4 f = ((const float4*)bp)[t];
      Bt[bcol + t * 4 + 0][brow] = (__bf16)f.x;
      Bt[bcol + t * 4 + 1][brow] = (__bf16)f.y;
      Bt[bcol + t * 4 + 2][brow] = (__bf16)f.z;
      Bt[bcol + t * 4 + 3][brow] = (__bf16)f.w;
    }
    if (k0 + BK < FDIM)
      __builtin_prefetch(ap + BK, 0, 0);
    __syncthreads();

    v16bf afrag[2], bfrag[2];
    #pragma unroll
    for (int mt = 0; mt < 2; ++mt) {
      union { v16bf v; v8bf h[2]; } u;
      const int m = wm + mt * 16 + l16;
      u.h[0] = *(const v8bf*)&As[m][kA];
      u.h[1] = *(const v8bf*)&As[m][kA + 16];
      afrag[mt] = u.v;
    }
    #pragma unroll
    for (int nt = 0; nt < 2; ++nt) {
      union { v16bf v; v8bf h[2]; } u;
      const int n = wn + nt * 16 + l16;
      u.h[0] = *(const v8bf*)&Bt[n][kB];
      u.h[1] = *(const v8bf*)&Bt[n][kB + 16];
      bfrag[nt] = u.v;
    }
    #pragma unroll
    for (int mt = 0; mt < 2; ++mt)
      #pragma unroll
      for (int nt = 0; nt < 2; ++nt)
        acc[mt][nt] = __builtin_amdgcn_wmma_f32_16x16x32_bf16(
            false, afrag[mt], false, bfrag[nt], (short)0, acc[mt][nt],
            false, false);
    __syncthreads();
  }

  #pragma unroll
  for (int mt = 0; mt < 2; ++mt)
    #pragma unroll
    for (int nt = 0; nt < 2; ++nt)
      #pragma unroll
      for (int v = 0; v < 8; ++v) {
        const int row = m0 + wm + mt * 16 + ((lane < 16) ? v : v + 8);
        const int col = n0 + wn + nt * 16 + l16;
        const float val = acc[mt][nt][v];
        Wh[(size_t)row * FDIM + col]  = val;
        Whb[(size_t)row * FDIM + col] = (__bf16)val;
      }
}

// ---------------------------------------------------------------------------
// K2: s1[i] = Wh[i,:]·a1 ; s2[i] = Wh[i,:]·a2  (one wave32 per row)
// ---------------------------------------------------------------------------
__global__ __launch_bounds__(256) void k_rowdot(const float* __restrict__ Wh,
                                                const float* __restrict__ a,
                                                float* __restrict__ s1,
                                                float* __restrict__ s2) {
  const int row  = blockIdx.x * 8 + (threadIdx.x >> 5);
  const int lane = threadIdx.x & 31;
  const float* r = Wh + (size_t)row * FDIM;
  float a1 = 0.f, a2 = 0.f;
  for (int f = lane; f < FDIM; f += 32) {
    const float w = r[f];
    a1 += w * a[f];
    a2 += w * a[FDIM + f];
  }
  a1 = waveReduceSum(a1);
  a2 = waveReduceSum(a2);
  if (lane == 0) { s1[row] = a1; s2[row] = a2; }
}

// ---------------------------------------------------------------------------
// K3: masked leaky-relu logits + row softmax -> attention (f32) to d_out.
// One 256-thread block per row, row cached in 32KB LDS.
// ---------------------------------------------------------------------------
__global__ __launch_bounds__(256) void k_softmax(const int* __restrict__ adj,
                                                 const float* __restrict__ s1,
                                                 const float* __restrict__ s2,
                                                 float* __restrict__ outAtt) {
  __shared__ float rowv[N_NODES];   // 32 KB
  __shared__ float redm[8];
  __shared__ float reds[8];

  const int i    = blockIdx.x;
  const int tid  = threadIdx.x;
  const int wv   = tid >> 5;
  const int lane = tid & 31;
  const float s1i = s1[i];
  const int* arow = adj + (size_t)i * N_NODES;

  float tmax = -INFINITY;
  for (int j = tid; j < N_NODES; j += 256) {
    float x = s1i + s2[j];
    x = (x > 0.f) ? x : ALPHA * x;
    const float v = (arow[j] > 0) ? x : MASKV;
    rowv[j] = v;
    tmax = fmaxf(tmax, v);
  }
  tmax = waveReduceMax(tmax);
  if (lane == 0) redm[wv] = tmax;
  __syncthreads();
  if (tid == 0) {
    float m = redm[0];
    #pragma unroll
    for (int k = 1; k < 8; ++k) m = fmaxf(m, redm[k]);
    redm[0] = m;
  }
  __syncthreads();
  const float rmax = redm[0];

  float tsum = 0.f;
  for (int j = tid; j < N_NODES; j += 256) {
    const float e = __expf(rowv[j] - rmax);
    rowv[j] = e;
    tsum += e;
  }
  tsum = waveReduceSum(tsum);
  if (lane == 0) reds[wv] = tsum;
  __syncthreads();
  if (tid == 0) {
    float s = 0.f;
    #pragma unroll
    for (int k = 0; k < 8; ++k) s += reds[k];
    reds[0] = s;
  }
  __syncthreads();
  const float inv = 1.0f / reds[0];

  float* orow = outAtt + (size_t)i * N_NODES;
  for (int j = tid; j < N_NODES; j += 256)
    orow[j] = rowv[j] * inv;
}

// ---------------------------------------------------------------------------
// K4: h_prime = attention (8192x8192 f32, from d_out) @ Wh (bf16), ELU, f32 out
// ---------------------------------------------------------------------------
__global__ __launch_bounds__(256) void k_gemm_att(const float* __restrict__ Att,
                                                  const __bf16* __restrict__ Whb,
                                                  float* __restrict__ Out) {
  __shared__ __bf16 As[BM][LDS_STRIDE];
  __shared__ __bf16 Bt[BN][LDS_STRIDE];

  const int tid  = threadIdx.x;
  const int m0   = blockIdx.x * BM;
  const int n0   = blockIdx.y * BN;
  const int wv   = tid >> 5;
  const int lane = tid & 31;
  const int wm   = (wv & 3) * 32;
  const int wn   = (wv >> 2) * 32;
  const int l16  = lane & 15;
  const int kA   = (lane >> 4) * 8;
  const int kB   = (lane >> 4) * 8;

  const int arow = tid >> 1;
  const int acol = (tid & 1) * 16;
  const int brow = tid >> 3;
  const int bcol = (tid & 7) * 8;

  v8f acc[2][2] = {};

  for (int k0 = 0; k0 < N_NODES; k0 += BK) {
    const float* ap = Att + (size_t)(m0 + arow) * N_NODES + (k0 + acol);
    #pragma unroll
    for (int t = 0; t < 4; ++t) {
      float4 f = ((const float4*)ap)[t];
      As[arow][acol + t * 4 + 0] = (__bf16)f.x;
      As[arow][acol + t * 4 + 1] = (__bf16)f.y;
      As[arow][acol + t * 4 + 2] = (__bf16)f.z;
      As[arow][acol + t * 4 + 3] = (__bf16)f.w;
    }
    const v8bf bvec = *(const v8bf*)(Whb + (size_t)(k0 + brow) * FDIM + (n0 + bcol));
    #pragma unroll
    for (int e = 0; e < 8; ++e)
      Bt[bcol + e][brow] = bvec[e];
    if (k0 + BK < N_NODES)
      __builtin_prefetch(ap + BK, 0, 0);   // global_prefetch_b8: next A tile
    __syncthreads();

    v16bf afrag[2], bfrag[2];
    #pragma unroll
    for (int mt = 0; mt < 2; ++mt) {
      union { v16bf v; v8bf h[2]; } u;
      const int m = wm + mt * 16 + l16;
      u.h[0] = *(const v8bf*)&As[m][kA];
      u.h[1] = *(const v8bf*)&As[m][kA + 16];
      afrag[mt] = u.v;
    }
    #pragma unroll
    for (int nt = 0; nt < 2; ++nt) {
      union { v16bf v; v8bf h[2]; } u;
      const int n = wn + nt * 16 + l16;
      u.h[0] = *(const v8bf*)&Bt[n][kB];
      u.h[1] = *(const v8bf*)&Bt[n][kB + 16];
      bfrag[nt] = u.v;
    }
    #pragma unroll
    for (int mt = 0; mt < 2; ++mt)
      #pragma unroll
      for (int nt = 0; nt < 2; ++nt)
        acc[mt][nt] = __builtin_amdgcn_wmma_f32_16x16x32_bf16(
            false, afrag[mt], false, bfrag[nt], (short)0, acc[mt][nt],
            false, false);
    __syncthreads();
  }

  #pragma unroll
  for (int mt = 0; mt < 2; ++mt)
    #pragma unroll
    for (int nt = 0; nt < 2; ++nt)
      #pragma unroll
      for (int v = 0; v < 8; ++v) {
        const int row = m0 + wm + mt * 16 + ((lane < 16) ? v : v + 8);
        const int col = n0 + wn + nt * 16 + l16;
        const float x = acc[mt][nt][v];
        Out[(size_t)row * FDIM + col] = (x > 0.f) ? x : (__expf(x) - 1.0f);
      }
}

// ---------------------------------------------------------------------------
extern "C" void kernel_launch(void* const* d_in, const int* in_sizes, int n_in,
                              void* d_out, int out_size, void* d_ws, size_t ws_size,
                              hipStream_t stream) {
  const float* h   = (const float*)d_in[0];   // (8192, 512)
  const int*   adj = (const int*)d_in[1];     // (8192, 8192)
  const float* W   = (const float*)d_in[2];   // (512, 512)
  const float* a   = (const float*)d_in[3];   // (1024, 1)

  float*  outElu = (float*)d_out;                         // 8192*512
  float*  outAtt = outElu + (size_t)N_NODES * FDIM;       // 8192*8192

  float*  Wh  = (float*)d_ws;                             // 16 MB
  __bf16* Whb = (__bf16*)(Wh + (size_t)N_NODES * FDIM);   // 8 MB
  float*  s1  = (float*)(Whb + (size_t)N_NODES * FDIM);
  float*  s2  = s1 + N_NODES;

  // K1: Wh = h @ W (f32 + bf16 copies)
  k_gemm_wh<<<dim3(N_NODES / BM, FDIM / BN), 256, 0, stream>>>(h, W, Wh, Whb);
  // K2: s1/s2 projections
  k_rowdot<<<N_NODES / 8, 256, 0, stream>>>(Wh, a, s1, s2);
  // K3: masked softmax -> attention (second output, also A of next GEMM)
  k_softmax<<<N_NODES, 256, 0, stream>>>(adj, s1, s2, outAtt);
  // K4: h_prime = attention @ Wh, ELU -> first output
  k_gemm_att<<<dim3(N_NODES / BM, FDIM / BN), 256, 0, stream>>>(outAtt, Whb, outElu);
}
